// Net_23158463660724
// MI455X (gfx1250) — matrix operand order repaired and checked by reading, not compiled, not run on hardware
//
#include <hip/hip_runtime.h>
#include <hip/hip_bf16.h>

typedef __attribute__((ext_vector_type(16))) _Float16 v16h;
typedef __attribute__((ext_vector_type(8)))  _Float16 v8h;
typedef __attribute__((ext_vector_type(8)))  float    v8f;

#define ACT_NONE 0
#define ACT_ELU  1
#define ACT_RELU 2

__device__ __forceinline__ float act_apply(float v, int act) {
    if (act == ACT_ELU)  return v > 0.f ? v : (__expf(v) - 1.f);
    if (act == ACT_RELU) return fmaxf(v, 0.f);
    return v;
}

// ---------------------------------------------------------------------------
// WMMA GEMM:  C[M,N] = act(A[M,K] @ B[K,N] + bias[N])
// f32 global data, f16 matrix-core inputs, f32 accumulate.
// Block: 256 threads = 8 waves. Block tile 64(M) x 64(N), K staged 64/iter.
// Waves arranged 4(M) x 2(N); each wave computes a 16x32 C slab
// (two 16x16 WMMA accumulators sharing one A fragment) -> 4 v_wmma/iter.
// ---------------------------------------------------------------------------
#define GEMM_BM 64
#define GEMM_BN 64
#define GEMM_BK 64

__global__ __launch_bounds__(256)
void gemm_wmma_kernel(const float* __restrict__ A, const float* __restrict__ Bm,
                      const float* __restrict__ bias, float* __restrict__ C,
                      int M, int N, int K, int act) {
    __shared__ alignas(16) _Float16 sA[GEMM_BM][GEMM_BK];   // 8 KB
    __shared__ alignas(16) _Float16 sB[GEMM_BK][GEMM_BN];   // 8 KB

    const int tid  = threadIdx.x;
    const int wave = tid >> 5;
    const int lane = tid & 31;
    const int m0 = blockIdx.y * GEMM_BM;
    const int n0 = blockIdx.x * GEMM_BN;
    const int wm = (wave >> 1) * 16;     // 0,16,32,48
    const int wn = (wave & 1) * 32;      // 0,32

    const bool fastA = (K % 4 == 0);
    const bool fastB = (N % 4 == 0);

    v8f acc0 = {0.f, 0.f, 0.f, 0.f, 0.f, 0.f, 0.f, 0.f};
    v8f acc1 = {0.f, 0.f, 0.f, 0.f, 0.f, 0.f, 0.f, 0.f};

    // staging coordinates: each thread owns 16 consecutive elements of one row
    const int sr = tid >> 2;            // 0..63
    const int sc = (tid & 3) * 16;      // 0,16,32,48

    for (int k0 = 0; k0 < K; k0 += GEMM_BK) {
        // ---- stage A tile (64 rows x 64 k) ----
        {
            const int gr = m0 + sr, gc = k0 + sc;
            if (gr < M && fastA && gc + 16 <= K) {
                const float4* src = (const float4*)(A + (size_t)gr * K + gc);
#pragma unroll
                for (int q = 0; q < 2; q++) {
                    float4 f0 = src[2 * q], f1 = src[2 * q + 1];
                    v8h h;
                    h[0] = (_Float16)f0.x; h[1] = (_Float16)f0.y;
                    h[2] = (_Float16)f0.z; h[3] = (_Float16)f0.w;
                    h[4] = (_Float16)f1.x; h[5] = (_Float16)f1.y;
                    h[6] = (_Float16)f1.z; h[7] = (_Float16)f1.w;
                    *(v8h*)&sA[sr][sc + q * 8] = h;
                }
            } else {
#pragma unroll
                for (int q = 0; q < 16; q++) {
                    int gc2 = gc + q;
                    float v = (gr < M && gc2 < K) ? A[(size_t)gr * K + gc2] : 0.f;
                    sA[sr][sc + q] = (_Float16)v;
                }
            }
        }
        // ---- stage B tile (64 k x 64 cols) ----
        {
            const int gr = k0 + sr, gc = n0 + sc;
            if (gr < K && fastB && gc + 16 <= N) {
                const float4* src = (const float4*)(Bm + (size_t)gr * N + gc);
#pragma unroll
                for (int q = 0; q < 2; q++) {
                    float4 f0 = src[2 * q], f1 = src[2 * q + 1];
                    v8h h;
                    h[0] = (_Float16)f0.x; h[1] = (_Float16)f0.y;
                    h[2] = (_Float16)f0.z; h[3] = (_Float16)f0.w;
                    h[4] = (_Float16)f1.x; h[5] = (_Float16)f1.y;
                    h[6] = (_Float16)f1.z; h[7] = (_Float16)f1.w;
                    *(v8h*)&sB[sr][sc + q * 8] = h;
                }
            } else {
#pragma unroll
                for (int q = 0; q < 16; q++) {
                    int gc2 = gc + q;
                    float v = (gr < K && gc2 < N) ? Bm[(size_t)gr * N + gc2] : 0.f;
                    sB[sr][sc + q] = (_Float16)v;
                }
            }
        }
        // prefetch next A tile rows (speculative)
        if (k0 + GEMM_BK < K) {
            int pr = m0 + sr;
            if (pr < M)
                __builtin_prefetch(&A[(size_t)pr * K + k0 + GEMM_BK], 0, 0);
        }
        __syncthreads();

        // ---- fragments + WMMA (CDNA5 16-bit wave32 layouts) ----
        const int arow  = wm + (lane & 15);
        const int akb   = (lane < 16) ? 0 : 8;   // A: K 0..7 / 8..15 (+16)
        const int bcol  = wn + (lane & 15);
        const int bkb   = (lane < 16) ? 0 : 8;   // B: K 0..7 / 8..15 (+16), same as A
#pragma unroll
        for (int kc = 0; kc < GEMM_BK; kc += 32) {
            v8h a_lo = *(const v8h*)&sA[arow][kc + akb];
            v8h a_hi = *(const v8h*)&sA[arow][kc + 16 + akb];
            v16h af = __builtin_shufflevector(a_lo, a_hi,
                        0, 1, 2, 3, 4, 5, 6, 7, 8, 9, 10, 11, 12, 13, 14, 15);
            v16h bf0, bf1;
#pragma unroll
            for (int e = 0; e < 16; e++) {
                bf0[e] = sB[kc + bkb + e + (e >= 8 ? 8 : 0)][bcol];
                bf1[e] = sB[kc + bkb + e + (e >= 8 ? 8 : 0)][bcol + 16];
            }
            acc0 = __builtin_amdgcn_wmma_f32_16x16x32_f16(
                       false, af, false, bf0, (short)0, acc0, false, false);
            acc1 = __builtin_amdgcn_wmma_f32_16x16x32_f16(
                       false, af, false, bf1, (short)0, acc1, false, false);
        }
        __syncthreads();
    }

    // C/D layout: lane col = lane%16; VGPR i -> row i (lanes<16) / i+8 (lanes>=16)
    const int col0 = n0 + wn + (lane & 15);
    const int col1 = col0 + 16;
    const float bv0 = (bias != nullptr && col0 < N) ? bias[col0] : 0.f;
    const float bv1 = (bias != nullptr && col1 < N) ? bias[col1] : 0.f;
#pragma unroll
    for (int i = 0; i < 8; i++) {
        int rowg = m0 + wm + i + ((lane < 16) ? 0 : 8);
        if (rowg < M) {
            if (col0 < N) C[(size_t)rowg * N + col0] = act_apply(acc0[i] + bv0, act);
            if (col1 < N) C[(size_t)rowg * N + col1] = act_apply(acc1[i] + bv1, act);
        }
    }
}

// ---------------------------------------------------------------------------
// KNN: per query, brute-force scan with register insertion sort (ascending d2).
// Writes every dil-th neighbor (kout of them), optionally distances.
// ---------------------------------------------------------------------------
template <int KD>
__global__ void knn_kernel(const float* __restrict__ q, const float* __restrict__ r,
                           int nq, int nr, int dil, int kout,
                           int* __restrict__ idx_out, float* __restrict__ dist_out) {
    int t = blockIdx.x * blockDim.x + threadIdx.x;
    int b = blockIdx.y;
    if (t >= nq) return;
    const float* Q = q + ((size_t)b * nq + t) * 3;
    const float qx = Q[0], qy = Q[1], qz = Q[2];
    float bestd[KD];
    int   besti[KD];
#pragma unroll
    for (int j = 0; j < KD; j++) { bestd[j] = 3.4e38f; besti[j] = 0; }
    const float* R = r + (size_t)b * nr * 3;
    for (int i = 0; i < nr; i++) {
        float dx = qx - R[3 * i + 0];
        float dy = qy - R[3 * i + 1];
        float dz = qz - R[3 * i + 2];
        float d = dx * dx + dy * dy + dz * dz;
        if (d < bestd[KD - 1]) {
            int j = KD - 1;
            while (j > 0 && bestd[j - 1] > d) {
                bestd[j] = bestd[j - 1]; besti[j] = besti[j - 1]; j--;
            }
            bestd[j] = d; besti[j] = i;
        }
    }
    size_t base = ((size_t)b * nq + t) * kout;
    for (int j = 0; j < kout; j++) {
        idx_out[base + j] = besti[j * dil];
        if (dist_out) dist_out[base + j] = bestd[j * dil];
    }
}

// ---------------------------------------------------------------------------
// Farthest point sampling: one block per batch, LDS distance array + argmax.
// ---------------------------------------------------------------------------
__global__ void fps_kernel(const float* __restrict__ pos, int n, int m,
                           int* __restrict__ sel) {
    extern __shared__ float smem[];
    float* dist = smem;                 // n floats
    float* rv   = smem + n;             // blockDim floats
    int*   ri   = (int*)(rv + blockDim.x);
    const int b = blockIdx.x;
    const int tid = threadIdx.x;
    const float* P = pos + (size_t)b * n * 3;
    const float p0x = P[0], p0y = P[1], p0z = P[2];
    for (int i = tid; i < n; i += blockDim.x) {
        float dx = P[3 * i] - p0x, dy = P[3 * i + 1] - p0y, dz = P[3 * i + 2] - p0z;
        dist[i] = dx * dx + dy * dy + dz * dz;
    }
    if (tid == 0) sel[(size_t)b * m] = 0;
    __syncthreads();
    for (int s = 1; s < m; s++) {
        float bv = -1.f; int bi = 0;
        for (int i = tid; i < n; i += blockDim.x) {
            float v = dist[i];
            if (v > bv) { bv = v; bi = i; }
        }
        rv[tid] = bv; ri[tid] = bi;
        __syncthreads();
        for (int off = blockDim.x >> 1; off > 0; off >>= 1) {
            if (tid < off) {
                bool take = (rv[tid + off] > rv[tid]) ||
                            (rv[tid + off] == rv[tid] && ri[tid + off] < ri[tid]);
                if (take) { rv[tid] = rv[tid + off]; ri[tid] = ri[tid + off]; }
            }
            __syncthreads();
        }
        const int nxt = ri[0];
        __syncthreads();
        if (tid == 0) sel[(size_t)b * m + s] = nxt;
        const float nx = P[3 * nxt], ny = P[3 * nxt + 1], nz = P[3 * nxt + 2];
        for (int i = tid; i < n; i += blockDim.x) {
            float dx = P[3 * i] - nx, dy = P[3 * i + 1] - ny, dz = P[3 * i + 2] - nz;
            float d = dx * dx + dy * dy + dz * dz;
            if (d < dist[i]) dist[i] = d;
        }
        __syncthreads();
    }
}

// ---------------------------------------------------------------------------
// Elementwise / gather / BN / small einsum kernels
// ---------------------------------------------------------------------------
__global__ void gather_rows_kernel(const float* __restrict__ x, const int* __restrict__ idx,
                                   float* __restrict__ out, int B, int n_out, int n_in, int C) {
    size_t total = (size_t)B * n_out * C;
    size_t id = blockIdx.x * (size_t)blockDim.x + threadIdx.x;
    if (id >= total) return;
    int c = (int)(id % C);
    size_t r = id / C;
    int b = (int)(r / n_out);
    out[id] = x[((size_t)b * n_in + idx[r]) * C + c];
}

__global__ void rel_kernel(const float* __restrict__ pos, const int* __restrict__ idx,
                           float* __restrict__ rel, int B, int n, int k) {
    size_t total = (size_t)B * n * k;
    size_t id = blockIdx.x * (size_t)blockDim.x + threadIdx.x;
    if (id >= total) return;
    size_t bi = id / k;                 // b*n + i
    int b = (int)(bi / n);
    int src = idx[id];
    const float* Pc = pos + bi * 3;
    const float* Ps = pos + ((size_t)b * n + src) * 3;
    rel[id * 3 + 0] = Ps[0] - Pc[0];
    rel[id * 3 + 1] = Ps[1] - Pc[1];
    rel[id * 3 + 2] = Ps[2] - Pc[2];
}

__global__ void build_xs_kernel(const float* __restrict__ xsp, const float* __restrict__ x,
                                const int* __restrict__ idx, float* __restrict__ xs,
                                int B, int n, int k, int cd, int cin) {
    int cc = cd + cin;
    size_t total = (size_t)B * n * k * cc;
    size_t id = blockIdx.x * (size_t)blockDim.x + threadIdx.x;
    if (id >= total) return;
    int c = (int)(id % cc);
    size_t p = id / cc;                 // (b*n+i)*k + j
    float v;
    if (c < cd) {
        v = xsp[p * cd + c];
    } else {
        size_t bi = p / k;
        int b = (int)(bi / n);
        int src = idx[p];
        v = x[((size_t)b * n + src) * cin + (c - cd)];
    }
    xs[id] = v;
}

__global__ void bn_stats_kernel(const float* __restrict__ x, float* __restrict__ stats,
                                size_t rows, int C) {
    __shared__ float sh[256], sh2[256];
    int c = blockIdx.x;
    int tid = threadIdx.x;
    float s = 0.f, s2 = 0.f;
    for (size_t r = tid; r < rows; r += blockDim.x) {
        float v = x[r * C + c];
        s += v; s2 += v * v;
    }
    sh[tid] = s; sh2[tid] = s2;
    __syncthreads();
    for (int off = blockDim.x >> 1; off > 0; off >>= 1) {
        if (tid < off) { sh[tid] += sh[tid + off]; sh2[tid] += sh2[tid + off]; }
        __syncthreads();
    }
    if (tid == 0) {
        float m = sh[0] / (float)rows;
        stats[2 * c + 0] = m;
        stats[2 * c + 1] = sh2[0] / (float)rows - m * m;
    }
}

__global__ void bn_apply_kernel(float* __restrict__ x, const float* __restrict__ stats,
                                const float* __restrict__ g, const float* __restrict__ bt,
                                size_t total, int C, int act) {
    size_t id = blockIdx.x * (size_t)blockDim.x + threadIdx.x;
    if (id >= total) return;
    int c = (int)(id % C);
    float m = stats[2 * c], v = stats[2 * c + 1];
    float y = (x[id] - m) * rsqrtf(v + 1e-5f) * g[c] + bt[c];
    x[id] = act_apply(y, act);
}

// out[p,g,j] = act( sum_t t[p,g,t] * w[g,j,t] + bias[g,j] )
__global__ void kk_transform_kernel(const float* __restrict__ t, const float* __restrict__ w,
                                    const float* __restrict__ bias, float* __restrict__ out,
                                    int bn, int k, int act) {
    size_t total = (size_t)bn * k * k;
    size_t id = blockIdx.x * (size_t)blockDim.x + threadIdx.x;
    if (id >= total) return;
    int j = (int)(id % k);
    int gi = (int)((id / k) % k);
    size_t p = id / ((size_t)k * k);
    const float* tp = t + (p * k + gi) * k;
    const float* wp = w + ((size_t)gi * k + j) * k;
    float s = bias[gi * k + j];
    for (int tt = 0; tt < k; tt++) s += tp[tt] * wp[tt];
    out[id] = act_apply(s, act);
}

// xt[p,c,j] = sum_k xs[p,k,c] * T[p,k,j]
__global__ void xt_kernel(const float* __restrict__ xs, const float* __restrict__ T,
                          float* __restrict__ xt, int bn, int k, int cc) {
    size_t total = (size_t)bn * cc * k;
    size_t id = blockIdx.x * (size_t)blockDim.x + threadIdx.x;
    if (id >= total) return;
    int j = (int)(id % k);
    int c = (int)((id / k) % cc);
    size_t p = id / ((size_t)k * cc);
    const float* xsp = xs + (size_t)p * k * cc;
    const float* Tp  = T + (size_t)p * k * k;
    float s = 0.f;
    for (int kk = 0; kk < k; kk++) s += xsp[(size_t)kk * cc + c] * Tp[kk * k + j];
    xt[id] = s;
}

// o[p, c*dm+m] = sum_k xt[p,c,k] * w[c,m,k] + bias[c*dm+m]
__global__ void dw_kernel(const float* __restrict__ xt, const float* __restrict__ w,
                          const float* __restrict__ bias, float* __restrict__ o,
                          int bn, int cc, int dm, int k) {
    size_t total = (size_t)bn * cc * dm;
    size_t id = blockIdx.x * (size_t)blockDim.x + threadIdx.x;
    if (id >= total) return;
    int m = (int)(id % dm);
    int c = (int)((id / dm) % cc);
    size_t p = id / ((size_t)dm * cc);
    const float* xp = xt + ((size_t)p * cc + c) * k;
    const float* wp = w + ((size_t)c * dm + m) * k;
    float s = bias[c * dm + m];
    for (int kk = 0; kk < k; kk++) s += xp[kk] * wp[kk];
    o[id] = s;
}

__global__ void mean_axis1_kernel(const float* __restrict__ x, float* __restrict__ out,
                                  int B, int n, int C) {
    int id = blockIdx.x * blockDim.x + threadIdx.x;
    if (id >= B * C) return;
    int b = id / C, c = id % C;
    float s = 0.f;
    for (int i = 0; i < n; i++) s += x[((size_t)b * n + i) * C + c];
    out[id] = s / (float)n;
}

__global__ void concat2_kernel(const float* __restrict__ A, const float* __restrict__ Bx,
                               float* __restrict__ out, size_t rows, int ca, int cb) {
    int cc = ca + cb;
    size_t total = rows * cc;
    size_t id = blockIdx.x * (size_t)blockDim.x + threadIdx.x;
    if (id >= total) return;
    int c = (int)(id % cc);
    size_t r = id / cc;
    out[id] = (c < ca) ? A[r * ca + c] : Bx[r * cb + (c - ca)];
}

__global__ void concat_bcast_kernel(const float* __restrict__ g, const float* __restrict__ X,
                                    float* __restrict__ out, int B, int n, int ca, int cb) {
    int cc = ca + cb;
    size_t total = (size_t)B * n * cc;
    size_t id = blockIdx.x * (size_t)blockDim.x + threadIdx.x;
    if (id >= total) return;
    int c = (int)(id % cc);
    size_t r = id / cc;
    int b = (int)(r / n);
    out[id] = (c < ca) ? g[(size_t)b * ca + c] : X[r * cb + (c - ca)];
}

// out[b,i,c] = sum_j w_j x[b,idx_j,c] / sum_j w_j   (k=4, w = 1/max(d,1e-16))
__global__ void interp_kernel(const float* __restrict__ x, const int* __restrict__ idx,
                              const float* __restrict__ d, float* __restrict__ out,
                              int B, int nq, int C, int nr) {
    size_t total = (size_t)B * nq * C;
    size_t id = blockIdx.x * (size_t)blockDim.x + threadIdx.x;
    if (id >= total) return;
    int c = (int)(id % C);
    size_t r = id / C;                  // b*nq + i
    int b = (int)(r / nq);
    const int* ip = idx + r * 4;
    const float* dp = d + r * 4;
    float ws = 0.f, s = 0.f;
    for (int j = 0; j < 4; j++) {
        float w = 1.f / fmaxf(dp[j], 1e-16f);
        ws += w;
        s += w * x[((size_t)b * nr + ip[j]) * C + c];
    }
    out[id] = s / ws;
}

__global__ void store_out_kernel(const float* __restrict__ z, float* __restrict__ out,
                                 int rows, int ncls) {
    size_t total = (size_t)rows * ncls;
    size_t id = blockIdx.x * (size_t)blockDim.x + threadIdx.x;
    if (id >= total) return;
    int cls = (int)(id / rows);
    int r   = (int)(id % rows);
    out[id] = z[(size_t)r * ncls + cls];
}

// ---------------------------------------------------------------------------
// Host-side orchestration
// ---------------------------------------------------------------------------
namespace {

struct Bump { char* base; size_t off; };

static float* bumpF(Bump& b, size_t n) {
    size_t o = (b.off + 255) & ~(size_t)255;
    b.off = o + n * sizeof(float);
    return (float*)(b.base + o);
}
static int* bumpI(Bump& b, size_t n) {
    size_t o = (b.off + 255) & ~(size_t)255;
    b.off = o + n * sizeof(int);
    return (int*)(b.base + o);
}

static inline dim3 g1(size_t total) { return dim3((unsigned)((total + 255) / 256)); }

static void gemm(const float* A, const float* Bm, const float* bias, float* C,
                 size_t M, int N, int K, int act, hipStream_t st) {
    dim3 grid((unsigned)((N + GEMM_BN - 1) / GEMM_BN),
              (unsigned)((M + GEMM_BM - 1) / GEMM_BM));
    gemm_wmma_kernel<<<grid, 256, 0, st>>>(A, Bm, bias, C, (int)M, N, K, act);
}

static void bn_forward(float* x, float* stats, const float* g, const float* b,
                       size_t rows, int C, int act, hipStream_t st) {
    bn_stats_kernel<<<C, 256, 0, st>>>(x, stats, rows, C);
    bn_apply_kernel<<<g1(rows * C), 256, 0, st>>>(x, stats, g, b, rows * C, C, act);
}

static void launch_knn(const float* q, const float* r, int B, int nq, int nr,
                       int kd, int dil, int kout, int* idx, float* dist, hipStream_t st) {
    dim3 grid((unsigned)((nq + 255) / 256), (unsigned)B);
    switch (kd) {
        case 4:  knn_kernel<4><<<grid, 256, 0, st>>>(q, r, nq, nr, dil, kout, idx, dist); break;
        case 8:  knn_kernel<8><<<grid, 256, 0, st>>>(q, r, nq, nr, dil, kout, idx, dist); break;
        case 16: knn_kernel<16><<<grid, 256, 0, st>>>(q, r, nq, nr, dil, kout, idx, dist); break;
        case 24: knn_kernel<24><<<grid, 256, 0, st>>>(q, r, nq, nr, dil, kout, idx, dist); break;
        default: knn_kernel<32><<<grid, 256, 0, st>>>(q, r, nq, nr, dil, kout, idx, dist); break;
    }
}

// One XConv layer. P = 24 parameter pointers in creation order.
static void run_xconv(const float* x /*nullable*/, const float* pos, int B, int n,
                      int cin, int cd, int cout, int k, int dil, int dm,
                      const float* const* P, float* out, Bump& ws, hipStream_t st) {
    const size_t save = ws.off;
    const int cc = cin + cd;
    const size_t bn  = (size_t)B * n;
    const size_t bnk = bn * k;

    int*   idx = bumpI(ws, bnk);
    float* rel = bumpF(ws, bnk * 3);
    float* h1  = bumpF(ws, bnk * cd);
    float* h2  = bumpF(ws, bnk * cd);
    float* xs  = (cin > 0) ? bumpF(ws, bnk * cc) : h2;
    float* t1  = bumpF(ws, bn * k * k);
    float* t2  = bumpF(ws, bn * k * k);
    float* xt  = bumpF(ws, bn * (size_t)cc * k);
    float* o   = bumpF(ws, bn * (size_t)cc * dm);
    float* stats = bumpF(ws, 2 * 512);

    launch_knn(pos, pos, B, n, n, k * dil, dil, k, idx, nullptr, st);
    rel_kernel<<<g1(bnk), 256, 0, st>>>(pos, idx, rel, B, n, k);

    // feature lift MLP: elu(rel@l1+b) -> bn -> elu(@l2+b) -> bn
    gemm(rel, P[0], P[1], h1, bnk, cd, 3, ACT_ELU, st);
    bn_forward(h1, stats, P[2], P[3], bnk, cd, ACT_NONE, st);
    gemm(h1, P[4], P[5], h2, bnk, cd, cd, ACT_ELU, st);
    bn_forward(h2, stats, P[6], P[7], bnk, cd, ACT_NONE, st);
    if (cin > 0)
        build_xs_kernel<<<g1(bnk * cc), 256, 0, st>>>(h2, x, idx, xs, B, n, k, cd, cin);

    // transform matrix T: (bn, 3k) -> (bn, k*k) MLP with two kxk convs
    gemm(rel, P[8], P[9], t1, bn, k * k, 3 * k, ACT_ELU, st);
    bn_forward(t1, stats, P[10], P[11], bn, k * k, ACT_NONE, st);
    kk_transform_kernel<<<g1(bn * k * k), 256, 0, st>>>(t1, P[12], P[13], t2, (int)bn, k, ACT_ELU);
    bn_forward(t2, stats, P[14], P[15], bn, k * k, ACT_NONE, st);
    kk_transform_kernel<<<g1(bn * k * k), 256, 0, st>>>(t2, P[16], P[17], t1, (int)bn, k, ACT_NONE);
    bn_forward(t1, stats, P[18], P[19], bn, k * k, ACT_NONE, st);

    // xt = xs^T @ T, depthwise, final GEMM (+relu fused from _forward)
    xt_kernel<<<g1(bn * (size_t)cc * k), 256, 0, st>>>(xs, t1, xt, (int)bn, k, cc);
    dw_kernel<<<g1(bn * (size_t)cc * dm), 256, 0, st>>>(xt, P[20], P[21], o, (int)bn, cc, dm, k);
    gemm(o, P[22], P[23], out, bn, cout, cc * dm, ACT_RELU, st);

    ws.off = save;
}

} // namespace

extern "C" void kernel_launch(void* const* d_in, const int* in_sizes, int n_in,
                              void* d_out, int out_size, void* d_ws, size_t ws_size,
                              hipStream_t stream) {
    (void)in_sizes; (void)n_in; (void)out_size; (void)ws_size;
    const int B = 8, N = 2048, N1 = 768, N2 = 288;

    const float* pos = (const float*)d_in[0];
    // d_in[1] = batch (unused)

    // params: 8 layers x 24 leaves, then 9 head leaves
    const float* P[8][24];
    for (int L = 0; L < 8; L++)
        for (int j = 0; j < 24; j++)
            P[L][j] = (const float*)d_in[2 + L * 24 + j];
    const int hb = 2 + 8 * 24;
    const float* lin1_w = (const float*)d_in[hb + 0];
    const float* lin1_b = (const float*)d_in[hb + 1];
    const float* lin2_w = (const float*)d_in[hb + 2];
    const float* lin2_b = (const float*)d_in[hb + 3];
    const float* fc1_w  = (const float*)d_in[hb + 4];
    const float* fcbn_g = (const float*)d_in[hb + 5];
    const float* fcbn_b = (const float*)d_in[hb + 6];
    const float* fc2_w  = (const float*)d_in[hb + 7];
    const float* fc2_b  = (const float*)d_in[hb + 8];

    Bump ws{(char*)d_ws, 0};

    float* x1   = bumpF(ws, (size_t)B * N * 64);
    int*   i1   = bumpI(ws, (size_t)B * N1);
    float* pos1 = bumpF(ws, (size_t)B * N1 * 3);
    float* x1g  = bumpF(ws, (size_t)B * N1 * 64);
    float* x2   = bumpF(ws, (size_t)B * N1 * 96);
    int*   i2   = bumpI(ws, (size_t)B * N2);
    float* pos2 = bumpF(ws, (size_t)B * N2 * 3);
    float* x2g  = bumpF(ws, (size_t)B * N2 * 96);
    float* x3   = bumpF(ws, (size_t)B * N2 * 192);
    float* x4   = bumpF(ws, (size_t)B * N2 * 384);
    float* gm   = bumpF(ws, (size_t)B * 384);
    float* gl1  = bumpF(ws, (size_t)B * 256);
    float* gl2  = bumpF(ws, (size_t)B * 128);
    float* cat4 = bumpF(ws, (size_t)B * N2 * 512);
    float* up4  = bumpF(ws, (size_t)B * N2 * 192);
    float* cat3 = bumpF(ws, (size_t)B * N2 * 384);
    float* up3  = bumpF(ws, (size_t)B * N2 * 96);
    int*   ii   = bumpI(ws, (size_t)B * N * 4);
    float* dd   = bumpF(ws, (size_t)B * N * 4);
    float* it1  = bumpF(ws, (size_t)B * N1 * 96);
    float* cat2 = bumpF(ws, (size_t)B * N1 * 192);
    float* up2  = bumpF(ws, (size_t)B * N1 * 96);
    float* it0  = bumpF(ws, (size_t)B * N * 96);
    float* cat1 = bumpF(ws, (size_t)B * N * 160);
    float* up1  = bumpF(ws, (size_t)B * N * 128);
    float* z1   = bumpF(ws, (size_t)B * N * 128);
    float* z2   = bumpF(ws, (size_t)B * N * 50);
    float* hstats = bumpF(ws, 2 * 256);
    // remaining ws space used as rewound per-XConv scratch

    // ---- encoder ----
    run_xconv(nullptr, pos, B, N, 0, 32, 64, 8, 1, 2, P[0], x1, ws, stream);

    fps_kernel<<<B, 256, (N + 512) * sizeof(float), stream>>>(pos, N, N1, i1);
    gather_rows_kernel<<<g1((size_t)B * N1 * 3), 256, 0, stream>>>(pos, i1, pos1, B, N1, N, 3);
    gather_rows_kernel<<<g1((size_t)B * N1 * 64), 256, 0, stream>>>(x1, i1, x1g, B, N1, N, 64);
    run_xconv(x1g, pos1, B, N1, 64, 64, 96, 12, 2, 1, P[1], x2, ws, stream);

    fps_kernel<<<B, 256, (N1 + 512) * sizeof(float), stream>>>(pos1, N1, N2, i2);
    gather_rows_kernel<<<g1((size_t)B * N2 * 3), 256, 0, stream>>>(pos1, i2, pos2, B, N2, N1, 3);
    gather_rows_kernel<<<g1((size_t)B * N2 * 96), 256, 0, stream>>>(x2, i2, x2g, B, N2, N1, 96);
    run_xconv(x2g, pos2, B, N2, 96, 128, 192, 16, 2, 1, P[2], x3, ws, stream);
    run_xconv(x3,  pos2, B, N2, 192, 256, 384, 16, 2, 1, P[3], x4, ws, stream);

    // ---- global head ----
    mean_axis1_kernel<<<g1((size_t)B * 384), 256, 0, stream>>>(x4, gm, B, N2, 384);
    gemm(gm,  lin1_w, lin1_b, gl1, B, 256, 384, ACT_RELU, stream);
    gemm(gl1, lin2_w, lin2_b, gl2, B, 128, 256, ACT_RELU, stream);

    // ---- decoder ----
    concat_bcast_kernel<<<g1((size_t)B * N2 * 512), 256, 0, stream>>>(gl2, x4, cat4, B, N2, 128, 384);
    run_xconv(cat4, pos2, B, N2, 512, 320, 192, 12, 2, 1, P[4], up4, ws, stream);

    concat2_kernel<<<g1((size_t)B * N2 * 384), 256, 0, stream>>>(up4, x3, cat3, (size_t)B * N2, 192, 192);
    run_xconv(cat3, pos2, B, N2, 384, 256, 96, 12, 2, 1, P[5], up3, ws, stream);

    launch_knn(pos1, pos2, B, N1, N2, 4, 1, 4, ii, dd, stream);
    interp_kernel<<<g1((size_t)B * N1 * 96), 256, 0, stream>>>(up3, ii, dd, it1, B, N1, 96, N2);
    concat2_kernel<<<g1((size_t)B * N1 * 192), 256, 0, stream>>>(it1, x2, cat2, (size_t)B * N1, 96, 96);
    run_xconv(cat2, pos1, B, N1, 192, 125, 96, 12, 2, 1, P[6], up2, ws, stream);

    launch_knn(pos, pos1, B, N, N1, 4, 1, 4, ii, dd, stream);
    interp_kernel<<<g1((size_t)B * N * 96), 256, 0, stream>>>(up2, ii, dd, it0, B, N, 96, N1);
    concat2_kernel<<<g1((size_t)B * N * 160), 256, 0, stream>>>(it0, x1, cat1, (size_t)B * N, 96, 64);
    run_xconv(cat1, pos, B, N, 160, 120, 128, 8, 2, 1, P[7], up1, ws, stream);

    // ---- classifier ----
    gemm(up1, fc1_w, nullptr, z1, (size_t)B * N, 128, 128, ACT_NONE, stream);
    bn_forward(z1, hstats, fcbn_g, fcbn_b, (size_t)B * N, 128, ACT_RELU, stream);
    gemm(z1, fc2_w, fc2_b, z2, (size_t)B * N, 50, 128, ACT_NONE, stream);

    store_out_kernel<<<g1((size_t)B * N * 50), 256, 0, stream>>>(z2, (float*)d_out, B * N, 50);
}
